// GNNMolEncoder_85117661872323
// MI455X (gfx1250) — matrix operand order repaired and checked by reading, not compiled, not run on hardware
//
#include <hip/hip_runtime.h>

// ---------------------------------------------------------------------------
// GNN (GINE) encoder for MI455X / gfx1250, wave32, WMMA bf16.
// GEMMs: v_wmma_f32_16x16x32_bf16, 64x128 block tile, 4 WMMA per wave per
// K-step, async global->LDS staging for the weight tile when available.
// Message passing uses float4 loads + global float atomics.
// ---------------------------------------------------------------------------

typedef __attribute__((ext_vector_type(16))) __bf16 v16bf;
typedef __attribute__((ext_vector_type(8)))  float  v8f;

union Frag { v16bf v; unsigned u[8]; };

enum { Nn = 50000, Ee = 800000, Gg = 2048,
       ND = 64, ED = 16, H = 128, L = 5, FFN = 512, OUTD = 512,
       JK = H * (L + 1) };   // 768

#if defined(__AMDGCN__) && __has_builtin(__builtin_amdgcn_global_load_async_to_lds_b128)
#define USE_ASYNC_LDS 1
typedef int i32x4 __attribute__((vector_size(16)));
typedef __attribute__((address_space(1))) i32x4 gi32x4;
typedef __attribute__((address_space(3))) i32x4 li32x4;
#else
#define USE_ASYNC_LDS 0
#endif

__device__ __forceinline__ void wait_async0() {
#if defined(__AMDGCN__)
#if __has_builtin(__builtin_amdgcn_s_wait_asynccnt)
    __builtin_amdgcn_s_wait_asynccnt(0);
#else
    asm volatile("s_wait_asynccnt 0x0" ::: "memory");
#endif
#endif
}

__device__ __forceinline__ unsigned f2bf(float f) {
    unsigned u = __float_as_uint(f);
    return (u + 0x7fffu + ((u >> 16) & 1u)) >> 16;   // RNE truncate to bf16
}
__device__ __forceinline__ unsigned pack2(float a, float b) {
    return f2bf(a) | (f2bf(b) << 16);
}

// ---------------------------------------------------------------------------
// Tiled GEMM: C[M,Nc] = Aeff[M,K] * W[K,Nc] + bias
//   FUSED: Aeff = (1+*eps)*A + A2      (GINE: z = (1+eps)*h + aggr)
//   else : Aeff = A
// W pre-converted bf16, transposed: Wt[Nc][Kpad].
// Block tile 64(M) x 128(N); 8 waves as 4(m) x 2(n); each wave: 16 rows x
// 64 cols -> one A fragment feeds 4 v_wmma_f32_16x16x32_bf16 per K-step.
// ---------------------------------------------------------------------------
template <bool FUSED>
__global__ __launch_bounds__(256)
void gemm_bf16_kernel(const float* __restrict__ A, int lda,
                      const float* __restrict__ A2, int lda2,
                      const float* __restrict__ eps_ptr,
                      const unsigned short* __restrict__ Wt,   // [Nc][Kpad] bf16
                      const float* __restrict__ bias,
                      float* __restrict__ C, int ldc,
                      int M, int Kreal, int Kpad)
{
    __shared__ unsigned la[64][17];    // A tile:   [m][k-pair] (pad 17: conflict-free)
    __shared__ unsigned lb[128][17];   // B tile^T: [n][k-pair]

    const int tid  = threadIdx.x;
    const int m0   = blockIdx.x * 64;
    const int n0   = blockIdx.y * 128;

    float ascale = 1.0f;
    if (FUSED) ascale = 1.0f + *eps_ptr;

    // A staging: thread -> (row 0..63, 8-float K chunk 0..3)
    const int sm = tid >> 2;
    const int sq = (tid & 3) * 8;
    const int sp = sq >> 1;
    // B staging: thread -> (row 0..127, 32-byte half-chunk 0..1)
    const int rb = tid >> 1;
    const int hb = tid & 1;

    // compute mapping
    const int lane = tid & 31;
    const int l    = lane & 15;
    const int half = lane >> 4;
    const int mrow = 16 * ((tid >> 5) & 3);
    const int ncol = 64 * (tid >> 7);

    v8f acc[4] = {v8f{}, v8f{}, v8f{}, v8f{}};

    for (int k0 = 0; k0 < Kpad; k0 += 32) {
        // ---- stage B tile (bf16 weights, K-contiguous) ----
        {
            const unsigned short* pb = Wt + (size_t)(n0 + rb) * Kpad + k0 + hb * 16;
#if USE_ASYNC_LDS
            __builtin_amdgcn_global_load_async_to_lds_b128(
                (gi32x4*)pb, (li32x4*)&lb[rb][hb * 8], 0, 0);
            __builtin_amdgcn_global_load_async_to_lds_b128(
                (gi32x4*)(pb + 8), (li32x4*)&lb[rb][hb * 8 + 4], 0, 0);
#else
            const uint4 b0 = *reinterpret_cast<const uint4*>(pb);
            const uint4 b1 = *reinterpret_cast<const uint4*>(pb + 8);
            lb[rb][hb*8 + 0] = b0.x; lb[rb][hb*8 + 1] = b0.y;
            lb[rb][hb*8 + 2] = b0.z; lb[rb][hb*8 + 3] = b0.w;
            lb[rb][hb*8 + 4] = b1.x; lb[rb][hb*8 + 5] = b1.y;
            lb[rb][hb*8 + 6] = b1.z; lb[rb][hb*8 + 7] = b1.w;
#endif
        }
        // ---- stage A tile (fp32 -> bf16 pairs, optional GINE fusion) ----
        {
            const int gm = m0 + sm;
            const int gk = k0 + sq;
            float f[8];
            if (gm < M && gk < Kreal) {
                const float* pa = A + (size_t)gm * lda + gk;
                float4 v0 = *reinterpret_cast<const float4*>(pa);
                float4 v1 = *reinterpret_cast<const float4*>(pa + 4);
                f[0]=v0.x*ascale; f[1]=v0.y*ascale; f[2]=v0.z*ascale; f[3]=v0.w*ascale;
                f[4]=v1.x*ascale; f[5]=v1.y*ascale; f[6]=v1.z*ascale; f[7]=v1.w*ascale;
                if (FUSED) {
                    const float* p2 = A2 + (size_t)gm * lda2 + gk;
                    float4 a0 = *reinterpret_cast<const float4*>(p2);
                    float4 a1 = *reinterpret_cast<const float4*>(p2 + 4);
                    f[0]+=a0.x; f[1]+=a0.y; f[2]+=a0.z; f[3]+=a0.w;
                    f[4]+=a1.x; f[5]+=a1.y; f[6]+=a1.z; f[7]+=a1.w;
                }
                if (k0 + 32 < Kpad)   // prefetch next K chunk of the streamed operand
                    __builtin_prefetch(pa + 32, 0, 1);
            } else {
                #pragma unroll
                for (int i = 0; i < 8; ++i) f[i] = 0.0f;
            }
            #pragma unroll
            for (int i = 0; i < 4; ++i)
                la[sm][sp + i] = pack2(f[2*i], f[2*i+1]);
        }
#if USE_ASYNC_LDS
        wait_async0();
#endif
        __syncthreads();

        // ---- fragments: A and B share contiguous k order, k = 16*half + s ----
        Frag a, b[4];
        #pragma unroll
        for (int v = 0; v < 8; ++v) {
            const int kpA = 8 * half + v;
            const int kpB = 8 * half + v;
            a.u[v] = la[mrow + l][kpA];
            #pragma unroll
            for (int j = 0; j < 4; ++j)
                b[j].u[v] = lb[ncol + 16 * j + l][kpB];
        }
        #pragma unroll
        for (int j = 0; j < 4; ++j)
            acc[j] = __builtin_amdgcn_wmma_f32_16x16x32_bf16(false, a.v, false, b[j].v,
                                                             (short)0, acc[j], false, false);
        __syncthreads();
    }

    // ---- epilogue: C layout VGPR r, lane -> (M = r + 8*half, N = l) ----
    #pragma unroll
    for (int r = 0; r < 8; ++r) {
        const int row = m0 + mrow + r + 8 * half;
        if (row < M) {
            #pragma unroll
            for (int j = 0; j < 4; ++j) {
                const int cc = n0 + ncol + 16 * j + l;
                C[(size_t)row * ldc + cc] = acc[j][r] + bias[cc];
            }
        }
    }
}

// ---------------------------------------------------------------------------
// Weight convert + transpose: fp32 W[K][Nc] -> bf16 Wt[Nc][Kpad] (zero pad)
// ---------------------------------------------------------------------------
__global__ void cvt_wt_kernel(const float* __restrict__ W, unsigned short* __restrict__ Wt,
                              int K, int Nc, int Kpad)
{
    int i = blockIdx.x * blockDim.x + threadIdx.x;
    if (i >= Nc * Kpad) return;
    int n = i / Kpad, k = i - n * Kpad;
    Wt[i] = (k < K) ? (unsigned short)f2bf(W[(size_t)k * Nc + n]) : (unsigned short)0;
}

__global__ void zero_kernel(float* __restrict__ p, size_t n)
{
    size_t i = (size_t)blockIdx.x * blockDim.x + threadIdx.x;
    if (i < n) p[i] = 0.0f;
}

// ---------------------------------------------------------------------------
// Message + scatter: msg = relu(h[src] + ea); aggr[dst] += msg
// 32 lanes per edge, float4 per lane (H = 128).
// ---------------------------------------------------------------------------
__global__ __launch_bounds__(256)
void msg_scatter_kernel(const float* __restrict__ h, int ldh,
                        const float* __restrict__ ea,
                        const int* __restrict__ src, const int* __restrict__ dst,
                        float* __restrict__ aggr, int E)
{
    unsigned tid = blockIdx.x * 256u + threadIdx.x;
    unsigned e = tid >> 5;
    if (e >= (unsigned)E) return;
    int c = (int)(tid & 31u) * 4;
    int s = src[e], d = dst[e];
    float4 hv = *reinterpret_cast<const float4*>(h + (size_t)s * ldh + c);
    float4 ev = *reinterpret_cast<const float4*>(ea + (size_t)e * H + c);
    float* ap = aggr + (size_t)d * H + c;
    atomicAdd(ap + 0, fmaxf(hv.x + ev.x, 0.0f));
    atomicAdd(ap + 1, fmaxf(hv.y + ev.y, 0.0f));
    atomicAdd(ap + 2, fmaxf(hv.z + ev.z, 0.0f));
    atomicAdd(ap + 3, fmaxf(hv.w + ev.w, 0.0f));
}

// ---------------------------------------------------------------------------
// BatchNorm column stats: one thread per column, coalesced row sweep.
// ---------------------------------------------------------------------------
__global__ __launch_bounds__(256)
void col_stats_kernel(const float* __restrict__ y, int ldy, int M, int Nc,
                      float* __restrict__ sum, float* __restrict__ sumsq)
{
    int c = blockIdx.x * 256 + threadIdx.x;
    if (c >= Nc) return;
    int r0 = blockIdx.y * 1024;
    int r1 = r0 + 1024; if (r1 > M) r1 = M;
    float s = 0.0f, ss = 0.0f;
    for (int r = r0; r < r1; ++r) {
        float v = y[(size_t)r * ldy + c];
        s += v; ss += v * v;
    }
    atomicAdd(&sum[c], s);
    atomicAdd(&sumsq[c], ss);
}

__global__ void bn_relu_kernel(float* __restrict__ y, int ldy, int M, int Nc,
                               const float* __restrict__ sum, const float* __restrict__ sumsq,
                               const float* __restrict__ g, const float* __restrict__ be)
{
    int i = blockIdx.x * blockDim.x + threadIdx.x;
    if (i >= M * Nc) return;
    int r = i / Nc, c = i - r * Nc;
    float invM = 1.0f / (float)M;
    float mean = sum[c] * invM;
    float var  = sumsq[c] * invM - mean * mean;
    float sc   = g[c] * rsqrtf(var + 1e-5f);
    float v = (y[(size_t)r * ldy + c] - mean) * sc + be[c];
    y[(size_t)r * ldy + c] = v > 0.0f ? v : 0.0f;
}

// ---------------------------------------------------------------------------
// Global mean pool
// ---------------------------------------------------------------------------
__global__ __launch_bounds__(256)
void pool_sum_kernel(const float* __restrict__ Hc, const int* __restrict__ batch,
                     float* __restrict__ pool, int N)
{
    unsigned tid = blockIdx.x * 256u + threadIdx.x;
    unsigned n = tid / (JK / 4);
    if (n >= (unsigned)N) return;
    int c = (int)(tid % (JK / 4)) * 4;
    int g = batch[n];
    float4 v = *reinterpret_cast<const float4*>(Hc + (size_t)n * JK + c);
    float* pp = pool + (size_t)g * JK + c;
    atomicAdd(pp + 0, v.x); atomicAdd(pp + 1, v.y);
    atomicAdd(pp + 2, v.z); atomicAdd(pp + 3, v.w);
}

__global__ void pool_count_kernel(const int* __restrict__ batch, float* __restrict__ cnt, int N)
{
    int i = blockIdx.x * blockDim.x + threadIdx.x;
    if (i < N) atomicAdd(&cnt[batch[i]], 1.0f);
}

__global__ void pool_div_kernel(float* __restrict__ pool, const float* __restrict__ cnt, int total)
{
    int i = blockIdx.x * blockDim.x + threadIdx.x;
    if (i >= total) return;
    pool[i] /= fmaxf(cnt[i / JK], 1.0f);
}

// ---------------------------------------------------------------------------
static inline size_t align256(size_t x) { return (x + 255) & ~(size_t)255; }

extern "C" void kernel_launch(void* const* d_in, const int* in_sizes, int n_in,
                              void* d_out, int out_size, void* d_ws, size_t ws_size,
                              hipStream_t stream)
{
    (void)in_sizes; (void)n_in; (void)out_size; (void)ws_size;
    auto F = [&](int i) { return (const float*)d_in[i]; };

    // setup_inputs() dict insertion order, params flattened recursively:
    // 0:x 1:edge_attr 2:edge_index 3:batch 4:node_W 5:node_b 6:edge_W 7:edge_b
    // convs[i] (base 8+9i): eps,W1,b1,g1,be1,W2,b2,g2,be2
    // readout[j] (base 53+4j): W,b,g,be ; 61:final_W 62:final_b
    const float* x         = F(0);
    const float* edge_attr = F(1);
    const int*   edge_index= (const int*)d_in[2];
    const int*   batch     = (const int*)d_in[3];
    const float* node_W = F(4); const float* node_b = F(5);
    const float* edge_W = F(6); const float* edge_b = F(7);
    const float* final_W = F(61); const float* final_b = F(62);
    const int* src = edge_index;
    const int* dst = edge_index + Ee;

    // ---- workspace carve-up ----
    char* ws = (char*)d_ws; size_t off = 0;
    auto carve = [&](size_t bytes) -> char* { char* p = ws + off; off = align256(off + bytes); return p; };
    float* Hcat = (float*)carve((size_t)Nn * JK * 4);     // JK concat buffer (h lives in slices)
    float* ea   = (float*)carve((size_t)Ee * H * 4);
    float* aggr = (float*)carve((size_t)Nn * H * 4);
    float* y1   = (float*)carve((size_t)Nn * 2 * H * 4);
    float* pool = (float*)carve((size_t)Gg * JK * 4);
    float* cnt  = (float*)carve((size_t)Gg * 4);
    float* r0b  = (float*)carve((size_t)Gg * FFN * 4);
    float* r1b  = (float*)carve((size_t)Gg * FFN * 4);
    float* stats= (float*)carve(2 * 1024 * 4);
    unsigned short* nodeWt = (unsigned short*)carve((size_t)H * ND * 2);
    unsigned short* edgeWt = (unsigned short*)carve((size_t)H * 32 * 2);
    unsigned short* W1t[L]; unsigned short* W2t[L];
    for (int i = 0; i < L; ++i) {
        W1t[i] = (unsigned short*)carve((size_t)2 * H * H * 2);
        W2t[i] = (unsigned short*)carve((size_t)H * 2 * H * 2);
    }
    unsigned short* ro0t = (unsigned short*)carve((size_t)FFN * JK * 2);
    unsigned short* ro1t = (unsigned short*)carve((size_t)FFN * FFN * 2);
    unsigned short* fint = (unsigned short*)carve((size_t)OUTD * FFN * 2);

    auto cvt = [&](const float* W, unsigned short* Wt, int K, int Nc, int Kpad) {
        int tot = Nc * Kpad;
        cvt_wt_kernel<<<(tot + 255) / 256, 256, 0, stream>>>(W, Wt, K, Nc, Kpad);
    };
    auto gemm = [&](const float* A, int lda, const float* A2, int lda2, const float* epsp,
                    const unsigned short* Wt, const float* bias, float* C, int ldc,
                    int M, int Nc, int Kreal, int Kpad) {
        dim3 grid((unsigned)((M + 63) / 64), (unsigned)(Nc / 128));
        if (A2)
            gemm_bf16_kernel<true><<<grid, 256, 0, stream>>>(A, lda, A2, lda2, epsp, Wt, bias,
                                                             C, ldc, M, Kreal, Kpad);
        else
            gemm_bf16_kernel<false><<<grid, 256, 0, stream>>>(A, lda, A2, lda2, epsp, Wt, bias,
                                                              C, ldc, M, Kreal, Kpad);
    };
    auto stats_bn = [&](float* y, int ldy, int M, int Nc, const float* g, const float* be) {
        zero_kernel<<<(2 * Nc + 255) / 256, 256, 0, stream>>>(stats, (size_t)(2 * Nc));
        dim3 sg((unsigned)((Nc + 255) / 256), (unsigned)((M + 1023) / 1024));
        col_stats_kernel<<<sg, 256, 0, stream>>>(y, ldy, M, Nc, stats, stats + Nc);
        int tot = M * Nc;
        bn_relu_kernel<<<(tot + 255) / 256, 256, 0, stream>>>(y, ldy, M, Nc, stats, stats + Nc, g, be);
    };

    // ---- weight conversion (bf16, transposed, K zero-padded to 32) ----
    cvt(node_W, nodeWt, ND, H, ND);
    cvt(edge_W, edgeWt, ED, H, 32);
    for (int i = 0; i < L; ++i) {
        int base = 8 + 9 * i;
        cvt(F(base + 1), W1t[i], H, 2 * H, H);       // W1 [128,256]
        cvt(F(base + 5), W2t[i], 2 * H, H, 2 * H);   // W2 [256,128]
    }
    cvt(F(53), ro0t, JK, FFN, JK);
    cvt(F(57), ro1t, FFN, FFN, FFN);
    cvt(final_W, fint, FFN, OUTD, FFN);

    // ---- encoders ----
    gemm(x,         ND, nullptr, 0, nullptr, nodeWt, node_b, Hcat, JK, Nn, H, ND, ND);
    gemm(edge_attr, ED, nullptr, 0, nullptr, edgeWt, edge_b, ea,   H,  Ee, H, ED, 32);

    // ---- conv layers ----
    for (int i = 0; i < L; ++i) {
        int base = 8 + 9 * i;
        const float* epsp = F(base + 0);
        const float* b1 = F(base + 2); const float* g1 = F(base + 3); const float* be1 = F(base + 4);
        const float* b2 = F(base + 6); const float* g2 = F(base + 7); const float* be2 = F(base + 8);
        float* Hi = Hcat + i * H;
        float* Ho = Hcat + (i + 1) * H;

        zero_kernel<<<(unsigned)(((size_t)Nn * H + 255) / 256), 256, 0, stream>>>(aggr, (size_t)Nn * H);
        msg_scatter_kernel<<<(unsigned)(((size_t)Ee * 32 + 255) / 256), 256, 0, stream>>>(
            Hi, JK, ea, src, dst, aggr, Ee);

        // z = (1+eps)*h + aggr, fused into GEMM A-operand
        gemm(Hi, JK, aggr, H, epsp, W1t[i], b1, y1, 2 * H, Nn, 2 * H, H, H);
        stats_bn(y1, 2 * H, Nn, 2 * H, g1, be1);
        gemm(y1, 2 * H, nullptr, 0, nullptr, W2t[i], b2, Ho, JK, Nn, H, 2 * H, 2 * H);
        stats_bn(Ho, JK, Nn, H, g2, be2);
    }

    // ---- global mean pool ----
    zero_kernel<<<(unsigned)(((size_t)Gg * JK + 255) / 256), 256, 0, stream>>>(pool, (size_t)Gg * JK);
    zero_kernel<<<(Gg + 255) / 256, 256, 0, stream>>>(cnt, (size_t)Gg);
    pool_sum_kernel<<<(unsigned)(((size_t)Nn * (JK / 4) + 255) / 256), 256, 0, stream>>>(Hcat, batch, pool, Nn);
    pool_count_kernel<<<(Nn + 255) / 256, 256, 0, stream>>>(batch, cnt, Nn);
    pool_div_kernel<<<(Gg * JK + 255) / 256, 256, 0, stream>>>(pool, cnt, Gg * JK);

    // ---- readout ----
    gemm(pool, JK, nullptr, 0, nullptr, ro0t, F(54), r0b, FFN, Gg, FFN, JK, JK);
    stats_bn(r0b, FFN, Gg, FFN, F(55), F(56));
    gemm(r0b, FFN, nullptr, 0, nullptr, ro1t, F(58), r1b, FFN, Gg, FFN, FFN, FFN);
    stats_bn(r1b, FFN, Gg, FFN, F(59), F(60));
    gemm(r1b, FFN, nullptr, 0, nullptr, fint, final_b, (float*)d_out, OUTD, Gg, OUTD, FFN, FFN);
}
